// PlantLSTM_80925773791482
// MI455X (gfx1250) — matrix operand order repaired and checked by reading, not compiled, not run on hardware
//
#include <hip/hip_runtime.h>

// ---------------------------------------------------------------------------
// Persistent 2-layer LSTM with autoregressive feedback for gfx1250 (MI455X).
// Batch-partitioned: 16 WGs x 8 wave32 waves, each WG owns 16 batch rows
// (M=16 == WMMA M). Weights pre-packed to bf16 tile-linear layout, streamed
// from L2 (1.8MB resident). GEMMs use a bounded 4-tile (half-ktile) register
// software pipeline: 8 global_load_b128 in flight behind 4 back-to-back
// v_wmma ops, with buffer reuse in a rolled loop so the scheduler cannot
// over-hoist and spill. Activations live in LDS; LSTM cell state lives in
// WMMA accumulator registers per wave.
// ---------------------------------------------------------------------------

typedef __attribute__((ext_vector_type(16))) __bf16 v16bf;
typedef __attribute__((ext_vector_type(8)))  __bf16 v8bf;
typedef __attribute__((ext_vector_type(8)))  float  v8f;

union V16 { v16bf v; v8bf h[2]; };

constexpr int Bsz  = 256;
constexpr int Tlen = 512;
constexpr int H    = 256;
constexpr int NIN  = 32;
constexpr int NOUT = 8;
constexpr int EMB  = 32;

constexpr int LDK0 = 352;   // [x(32)|pv(8)|emb(32)|pad(24)|h0(256)]
constexpr int LDK1 = 512;   // [h0(256)|h1(256)]
constexpr int KT0  = 11;    // 352/32
constexpr int KT1  = 16;    // 512/32

// workspace byte offsets (weights packed as bf16 tile blocks of 1KB each)
constexpr size_t OFF_W0 = 0;                        // 64 ntiles * 11 ktiles
constexpr size_t OFF_W1 = OFF_W0 + 64*11*1024;      // 64 * 16
constexpr size_t OFF_F1 = OFF_W1 + 64*16*1024;      // 8 * 8
constexpr size_t OFF_F2 = OFF_F1 + 8*8*1024;        // 1 * 4
constexpr size_t OFF_B0 = OFF_F2 + 4*1024;          // 1024 f32
constexpr size_t OFF_B1 = OFF_B0 + 4096;            // 1024 f32

__device__ __forceinline__ v8f splat8(float x) {
  return (v8f){x, x, x, x, x, x, x, x};
}
__device__ __forceinline__ float sigm(float x) {
  return __builtin_amdgcn_rcpf(1.f + __builtin_amdgcn_exp2f(-1.44269504f * x));
}
__device__ __forceinline__ float tanhe(float x) {
  return 2.f * __builtin_amdgcn_rcpf(1.f + __builtin_amdgcn_exp2f(-2.88539008f * x)) - 1.f;
}

// A tile (16x32 bf16) slice for this lane, from a precomputed per-lane row
// pointer (LDS). Contiguous K layout: lanes 0-15: row=lane, K=kb..kb+15
// with kb=0; lanes 16-31: kb=16. byteoff = kt*64.
__device__ __forceinline__ v16bf loada2(const char* arow, int byteoff) {
  V16 r;
  r.h[0] = *(const v8bf*)(arow + byteoff);
  r.h[1] = *(const v8bf*)(arow + byteoff + 16);   // +8 bf16 elements
  return r.v;
}

// B tile (32x16 bf16) from tile-linear packed weights via per-lane base
// pointer + constant byte offset (folds into the global_load ioffset).
__device__ __forceinline__ v16bf loadb2(const char* p, int byteoff) {
  V16 r;
  r.h[0] = *(const v8bf*)(p + byteoff);
  r.h[1] = *(const v8bf*)(p + byteoff + 16);
  return r.v;
}

// generic A-tile load (used by FC head)
__device__ __forceinline__ v16bf load_a(const __bf16* buf, int ldk, int kbase, int lane) {
  int row = lane & 15;
  int kb  = kbase + ((lane >> 4) << 4);
  V16 r;
  r.h[0] = *(const v8bf*)(buf + row * ldk + kb);
  r.h[1] = *(const v8bf*)(buf + row * ldk + kb + 8);
  return r.v;
}
__device__ __forceinline__ v16bf load_b(const __bf16* w, int block, int lane) {
  const __bf16* p = w + (size_t)block * 512 + lane * 16;
  V16 r;
  r.h[0] = *(const v8bf*)(p);
  r.h[1] = *(const v8bf*)(p + 8);
  return r.v;
}

__device__ __forceinline__ v8f wmma_bf16(v16bf a, v16bf b, v8f c) {
  return __builtin_amdgcn_wmma_f32_16x16x32_bf16(false, a, false, b, (short)0,
                                                 c, false, false);
}

// 8 N-tiles per wave: gate g in {i,f,g,o}, unit-tile u in {0,1}:
// acc index i = g*2+u ; weight block = (g*16 + wave*2 + u)*KT + kt.
// Half-ktile (4-tile) software pipeline, kt loop rolled so the fixed
// bcur/bnxt buffers bound the live set (no spills, no over-hoisting).
template <int KT, int LDK>
__device__ __forceinline__ void gemm8(const __bf16* abuf, const __bf16* wbase,
                                      int wave, int lane, v8f acc[8]) {
  const char* arow = (const char*)(abuf + (lane & 15) * LDK + ((lane >> 4) << 4));
  const char* pw   = (const char*)wbase + (size_t)(wave * 2) * KT * 1024 + lane * 32;
  constexpr int GU = KT * 1024;        // byte stride between unit-tiles (u)
  constexpr int GG = 16 * KT * 1024;   // byte stride between gates (g)

  v16bf acur, anxt;
  v16bf bcur[4], bnxt[4];

  acur = loada2(arow, 0);
#pragma unroll
  for (int i = 0; i < 4; ++i)          // gates 0,1 (i,f) at kt=0
    bcur[i] = loadb2(pw, (i >> 1) * GG + (i & 1) * GU);

#pragma unroll 1
  for (int kt = 0; kt < KT; ++kt) {
    const int kb = kt * 1024;
#pragma unroll
    for (int i = 0; i < 4; ++i)        // prefetch gates 2,3 (g,o) at kt
      bnxt[i] = loadb2(pw, kb + ((i >> 1) + 2) * GG + (i & 1) * GU);
#pragma unroll
    for (int i = 0; i < 4; ++i)
      acc[i] = wmma_bf16(acur, bcur[i], acc[i]);
    if (kt + 1 < KT) {
      anxt = loada2(arow, (kt + 1) * 64);
#pragma unroll
      for (int i = 0; i < 4; ++i)      // prefetch gates 0,1 at kt+1
        bcur[i] = loadb2(pw, kb + 1024 + (i >> 1) * GG + (i & 1) * GU);
    }
#pragma unroll
    for (int i = 0; i < 4; ++i)
      acc[4 + i] = wmma_bf16(acur, bnxt[i], acc[4 + i]);
    acur = anxt;
  }
}

// ---------------------------------------------------------------------------
// One-shot weight packing: fp32 -> bf16 tile-linear blocks + fused biases.
// out[p] with p = block*512 + lane*16 + j ; n = ntile*16 + (lane&15),
// k = ktile*32 + (lane>>4)*16 + j.
// ---------------------------------------------------------------------------
__global__ void pack_kernel(const float* __restrict__ Wih0, const float* __restrict__ Whh0,
                            const float* __restrict__ bih0, const float* __restrict__ bhh0,
                            const float* __restrict__ Wih1, const float* __restrict__ Whh1,
                            const float* __restrict__ bih1, const float* __restrict__ bhh1,
                            const float* __restrict__ fc1w, const float* __restrict__ fc2w,
                            __bf16* __restrict__ w0, __bf16* __restrict__ w1,
                            __bf16* __restrict__ wf1, __bf16* __restrict__ wf2,
                            float* __restrict__ b0, float* __restrict__ b1) {
  int p = blockIdx.x * blockDim.x + threadIdx.x;
  const int N_W0 = 64 * 11 * 512, N_W1 = 64 * 16 * 512, N_F1 = 8 * 8 * 512, N_F2 = 4 * 512;

  if (p < N_W0) {
    int j = p & 15, lane = (p >> 4) & 31, block = p >> 9;
    int ntile = block / 11, ktile = block - ntile * 11;
    int n = ntile * 16 + (lane & 15);
    int k = ktile * 32 + ((lane >> 4) << 4) + j;
    float v = 0.f;
    if (k < 72)       v = Wih0[n * 72 + k];
    else if (k >= 96) v = Whh0[n * 256 + (k - 96)];
    w0[p] = (__bf16)v;
    return;
  }
  p -= N_W0;
  if (p < N_W1) {
    int j = p & 15, lane = (p >> 4) & 31, block = p >> 9;
    int ntile = block >> 4, ktile = block & 15;
    int n = ntile * 16 + (lane & 15);
    int k = ktile * 32 + ((lane >> 4) << 4) + j;
    float v = (k < 256) ? Wih1[n * 256 + k] : Whh1[n * 256 + (k - 256)];
    w1[p] = (__bf16)v;
    return;
  }
  p -= N_W1;
  if (p < N_F1) {
    int j = p & 15, lane = (p >> 4) & 31, block = p >> 9;
    int ntile = block >> 3, ktile = block & 7;
    int n = ntile * 16 + (lane & 15);
    int k = ktile * 32 + ((lane >> 4) << 4) + j;
    wf1[p] = (__bf16)fc1w[n * 256 + k];
    return;
  }
  p -= N_F1;
  if (p < N_F2) {
    int j = p & 15, lane = (p >> 4) & 31, block = p >> 9;  // block = ktile 0..3
    int n = lane & 15;
    int k = block * 32 + ((lane >> 4) << 4) + j;
    wf2[p] = (__bf16)((n < 8) ? fc2w[n * 128 + k] : 0.f);
    return;
  }
  p -= N_F2;
  if (p < 1024) { b0[p] = bih0[p] + bhh0[p]; return; }
  p -= 1024;
  if (p < 1024) { b1[p] = bih1[p] + bhh1[p]; }
}

// ---------------------------------------------------------------------------
// Main persistent kernel: grid = 16 blocks x 256 threads (8 wave32 waves).
// ---------------------------------------------------------------------------
__global__ __launch_bounds__(256) void lstm_main(
    const float* __restrict__ x_cv, const float* __restrict__ pv_init,
    const int* __restrict__ scenario, const float* __restrict__ emb_table,
    const float* __restrict__ fc1_b, const float* __restrict__ fc2_b,
    const __bf16* __restrict__ w0, const __bf16* __restrict__ w1,
    const __bf16* __restrict__ wf1, const __bf16* __restrict__ wf2,
    const float* __restrict__ b0, const float* __restrict__ b1,
    float* __restrict__ out) {
  __shared__ __align__(16) __bf16 xbuf0[16 * LDK0];   // L0 A: x|pv|emb|pad|h0
  __shared__ __align__(16) __bf16 xbuf1[16 * LDK1];   // L1 A: h0|h1
  __shared__ __align__(16) __bf16 relu_s[16 * 128];   // FC2 A

  const int tid  = threadIdx.x;
  const int lane = tid & 31;
  const int wave = tid >> 5;
  const int m0   = blockIdx.x * 16;
  const int nl   = lane & 15;
  const int hi8  = (lane >> 4) << 3;

  for (int i = tid; i < 16 * LDK0; i += 256) xbuf0[i] = (__bf16)0.f;
  for (int i = tid; i < 16 * LDK1; i += 256) xbuf1[i] = (__bf16)0.f;
  __syncthreads();
  for (int i = tid; i < 16 * NOUT; i += 256) {
    int r = i >> 3, c = i & 7;
    xbuf0[r * LDK0 + 32 + c] = (__bf16)pv_init[(m0 + r) * NOUT + c];
  }
  for (int i = tid; i < 16 * EMB; i += 256) {
    int r = i >> 5, c = i & 31;
    xbuf0[r * LDK0 + 40 + c] = (__bf16)emb_table[scenario[m0 + r] * EMB + c];
  }

  // per-wave bias registers (loaded once)
  float bias0[8], bias1[8];
#pragma unroll
  for (int i = 0; i < 8; ++i) {
    int nt = (i >> 1) * 16 + (wave << 1) + (i & 1);
    bias0[i] = b0[nt * 16 + nl];
    bias1[i] = b1[nt * 16 + nl];
  }
  float biasf1 = fc1_b[wave * 16 + nl];
  float biasf2 = (nl < 8) ? fc2_b[nl] : 0.f;

  v8f c0[2] = {splat8(0.f), splat8(0.f)};   // layer-0 cell state (units 32w..32w+32)
  v8f c1[2] = {splat8(0.f), splat8(0.f)};   // layer-1 cell state

  __syncthreads();

#pragma unroll 1
  for (int t = 0; t < Tlen; ++t) {
    // stage x_t (cols 0..31); pv/emb/h0 regions untouched here
    for (int i = tid; i < 512; i += 256) {
      int r = i >> 5, c = i & 31;
      xbuf0[r * LDK0 + c] = (__bf16)x_cv[((m0 + r) * Tlen + t) * NIN + c];
    }
    __syncthreads();  // S1: x_t + pv(t-1) + h0(t-1) visible

    v8f acc[8];
#pragma unroll
    for (int i = 0; i < 8; ++i) acc[i] = splat8(bias0[i]);
    gemm8<KT0, LDK0>(xbuf0, w0, wave, lane, acc);
    __syncthreads();  // S2: all waves done reading xbuf0

    // layer-0 cell: combine i/f/g/o in registers, write h0 bf16 to LDS
#pragma unroll
    for (int u = 0; u < 2; ++u) {
#pragma unroll
      for (int e = 0; e < 8; ++e) {
        float iv = sigm(acc[u][e]);
        float fv = sigm(acc[2 + u][e]);
        float gv = tanhe(acc[4 + u][e]);
        float ov = sigm(acc[6 + u][e]);
        float c  = fv * c0[u][e] + iv * gv;
        c0[u][e] = c;
        float h  = ov * tanhe(c);
        int r    = e + hi8;
        int col  = (wave << 5) + (u << 4) + nl;
        __bf16 hb = (__bf16)h;
        xbuf1[r * LDK1 + col]      = hb;   // layer-1 input
        xbuf0[r * LDK0 + 96 + col] = hb;   // recurrent input for t+1
      }
    }
    __syncthreads();  // S3: h0(t) visible

#pragma unroll
    for (int i = 0; i < 8; ++i) acc[i] = splat8(bias1[i]);
    gemm8<KT1, LDK1>(xbuf1, w1, wave, lane, acc);
    __syncthreads();  // S4: all waves done reading xbuf1

#pragma unroll
    for (int u = 0; u < 2; ++u) {
#pragma unroll
      for (int e = 0; e < 8; ++e) {
        float iv = sigm(acc[u][e]);
        float fv = sigm(acc[2 + u][e]);
        float gv = tanhe(acc[4 + u][e]);
        float ov = sigm(acc[6 + u][e]);
        float c  = fv * c1[u][e] + iv * gv;
        c1[u][e] = c;
        float h  = ov * tanhe(c);
        int r    = e + hi8;
        int col  = (wave << 5) + (u << 4) + nl;
        xbuf1[r * LDK1 + 256 + col] = (__bf16)h;
      }
    }
    __syncthreads();  // S5: h1(t) visible

    // FC1: each wave one 16-col tile of relu(h1 @ fc1_w.T + b), pipelined
    {
      v8f af = splat8(biasf1);
      v16bf a0 = load_a(xbuf1, LDK1, 256, lane);
      v16bf bv = load_b(wf1, wave * 8, lane);
#pragma unroll
      for (int kt = 0; kt < 8; ++kt) {
        v16bf an = a0, bn = bv;
        if (kt < 7) {
          an = load_a(xbuf1, LDK1, 256 + (kt + 1) * 32, lane);
          bn = load_b(wf1, wave * 8 + kt + 1, lane);
        }
        af = wmma_bf16(a0, bv, af);
        a0 = an;
        bv = bn;
      }
#pragma unroll
      for (int e = 0; e < 8; ++e) {
        float v = af[e] > 0.f ? af[e] : 0.f;
        relu_s[(e + hi8) * 128 + (wave << 4) + nl] = (__bf16)v;
      }
    }
    __syncthreads();  // S6: relu output visible

    // FC2 (N=8 padded to 16): wave 0 only; writes output + pv feedback
    if (wave == 0) {
      v8f ao = splat8(biasf2);
      v16bf a0 = load_a(relu_s, 128, 0, lane);
      v16bf bv = load_b(wf2, 0, lane);
#pragma unroll
      for (int kt = 0; kt < 4; ++kt) {
        v16bf an = a0, bn = bv;
        if (kt < 3) {
          an = load_a(relu_s, 128, (kt + 1) * 32, lane);
          bn = load_b(wf2, kt + 1, lane);
        }
        ao = wmma_bf16(a0, bv, ao);
        a0 = an;
        bv = bn;
      }
      if (nl < 8) {
#pragma unroll
        for (int e = 0; e < 8; ++e) {
          int r = e + hi8;
          out[((m0 + r) * Tlen + t) * NOUT + nl] = ao[e];
          xbuf0[r * LDK0 + 32 + nl] = (__bf16)ao[e];   // autoregressive feedback
        }
      }
    }
    // next iteration's S1 orders wave0's pv write vs. all readers
  }
}

extern "C" void kernel_launch(void* const* d_in, const int* in_sizes, int n_in,
                              void* d_out, int out_size, void* d_ws, size_t ws_size,
                              hipStream_t stream) {
  const float* x_cv = (const float*)d_in[0];
  const float* pv   = (const float*)d_in[1];
  const int*   scen = (const int*)d_in[2];
  const float* embt = (const float*)d_in[3];
  const float* Wih0 = (const float*)d_in[4];
  const float* Whh0 = (const float*)d_in[5];
  const float* bih0 = (const float*)d_in[6];
  const float* bhh0 = (const float*)d_in[7];
  const float* Wih1 = (const float*)d_in[8];
  const float* Whh1 = (const float*)d_in[9];
  const float* bih1 = (const float*)d_in[10];
  const float* bhh1 = (const float*)d_in[11];
  const float* fc1w = (const float*)d_in[12];
  const float* fc1b = (const float*)d_in[13];
  const float* fc2w = (const float*)d_in[14];
  const float* fc2b = (const float*)d_in[15];

  char* ws = (char*)d_ws;
  __bf16* w0  = (__bf16*)(ws + OFF_W0);
  __bf16* w1  = (__bf16*)(ws + OFF_W1);
  __bf16* wf1 = (__bf16*)(ws + OFF_F1);
  __bf16* wf2 = (__bf16*)(ws + OFF_F2);
  float*  b0  = (float*)(ws + OFF_B0);
  float*  b1  = (float*)(ws + OFF_B1);

  // 64*11*512 + 64*16*512 + 8*8*512 + 4*512 + 1024 + 1024 = 921600 = 3600*256
  pack_kernel<<<3600, 256, 0, stream>>>(Wih0, Whh0, bih0, bhh0, Wih1, Whh1,
                                        bih1, bhh1, fc1w, fc2w, w0, w1, wf1,
                                        wf2, b0, b1);
  lstm_main<<<16, 256, 0, stream>>>(x_cv, pv, scen, embt, fc1b, fc2b, w0, w1,
                                    wf1, wf2, b0, b1, (float*)d_out);
}
